// PolicyNetRegression_11948599017634
// MI455X (gfx1250) — matrix operand order repaired and checked by reading, not compiled, not run on hardware
//
#include <hip/hip_runtime.h>
#include <hip/hip_bf16.h>
#include <math.h>

#define Bz   32
#define Sz   50
#define Lz   20
#define Vz   50000
#define Ez   128
#define Hz   256
#define NUSz 100
#define NUP  112   // NUS padded to multiple of 16

typedef __bf16 bf16;
typedef __attribute__((ext_vector_type(16))) __bf16 v16bf;
typedef __attribute__((ext_vector_type(8)))  __bf16 v8bf;
typedef __attribute__((ext_vector_type(8)))  float  v8f;
typedef __attribute__((ext_vector_type(4))) unsigned int v4u;
typedef __attribute__((ext_vector_type(8))) int          v8i;
typedef __attribute__((ext_vector_type(4))) int          v4i;

__device__ __forceinline__ float sigf(float x) { return 1.f / (1.f + __expf(-x)); }

// A fragment: 16x32 bf16 tile, row-major source, row stride = rstride elements.
// base points at (row0, k0). Layout: lanes 0-15 hold M=lane, K=0..15;
// lanes 16-31 hold M=lane-16, K=16..31.
__device__ __forceinline__ v16bf load_a_frag(const bf16* base, int rstride, int lane) {
  int m = lane & 15, half = lane >> 4;
  const bf16* p = base + m * rstride + half * 16;
  v8bf lo = *(const v8bf*)(p);        // K = half*16 + 0..7
  v8bf hi = *(const v8bf*)(p + 8);    // K = half*16 + 8..15
  v16bf a;
#pragma unroll
  for (int i = 0; i < 8; ++i) { a[i] = lo[i]; a[8 + i] = hi[i]; }
  return a;
}

// B fragment: 32x16 (KxN) from a row-major [N,K] weight matrix (i.e. B = W^T).
// Wk points at column k0 of W (W + k0). lanes 0-15: N=n0+lane, K=0..15;
// lanes 16-31: same N, K=16..31. One contiguous 32-byte load per lane.
__device__ __forceinline__ v16bf load_b_frag(const bf16* Wk, int ldk, int n0, int lane) {
  int n = n0 + (lane & 15), half = lane >> 4;
  return *(const v16bf*)(Wk + (size_t)n * ldk + half * 16);
}

// ---------------------------------------------------------------------------
// small utility kernels
// ---------------------------------------------------------------------------
__global__ void cvt_bf16_kernel(const float* __restrict__ src, bf16* __restrict__ dst, int n) {
  int i = blockIdx.x * blockDim.x + threadIdx.x;
  int stride = gridDim.x * blockDim.x;
  for (; i < n; i += stride) dst[i] = (bf16)src[i];
}

__global__ void zero_bf16_kernel(bf16* __restrict__ p, int n) {
  int i = blockIdx.x * blockDim.x + threadIdx.x;
  int stride = gridDim.x * blockDim.x;
  for (; i < n; i += stride) p[i] = (bf16)0.f;
}

__global__ void embed_kernel(const int* __restrict__ inputs, const int* __restrict__ mask,
                             const float* __restrict__ emb, bf16* __restrict__ xb) {
  int i = blockIdx.x * blockDim.x + threadIdx.x;
  int stride = gridDim.x * blockDim.x;
  const int total = Bz * Sz * Ez;
  for (; i < total; i += stride) {
    int e  = i & (Ez - 1);
    int bs = i >> 7;            // Ez == 128
    int id = inputs[bs] * mask[bs];
    xb[i] = (bf16)emb[(size_t)id * Ez + e];
  }
}

__global__ void util_init_kernel(const int* __restrict__ mask, int* __restrict__ last_idx,
                                 int* __restrict__ counters) {
  int t = threadIdx.x;
  if (t < Bz) {
    int s = 0;
    for (int j = 0; j < Sz; ++j) s += mask[t * Sz + j];
    last_idx[t] = s - 1;
  }
  if (t == Bz) { counters[0] = 0; counters[1] = 0; }
}

// ---------------------------------------------------------------------------
// LSTM layer: one workgroup (32 waves) runs the whole sequence.
// gates[32,1024] = X_t @ Wih^T + h @ Whh^T + bias, gate order i,f,g,o.
// Wave w owns output tile (mt = w>>4, nt = w&15): 16 batch rows x 16 H columns,
// computing all 4 gate tiles for those columns -> c state stays in registers.
// h is exchanged through LDS as bf16.
// ---------------------------------------------------------------------------
template <int DIN>
__global__ __launch_bounds__(1024) void lstm_layer_kernel(
    const bf16* __restrict__ x,
    const bf16* __restrict__ Wih, const bf16* __restrict__ Whh,
    const float* __restrict__ bih, const float* __restrict__ bhh,
    bf16* __restrict__ hseq, bf16* __restrict__ outb,
    const int* __restrict__ last_idx, int write_seq) {
  __shared__ bf16 hb[Bz * Hz];
  const int tid  = threadIdx.x;
  const int lane = tid & 31;
  const int w    = tid >> 5;
  const int nt   = w & 15;
  const int mt   = w >> 4;
  const int half = lane >> 4;
  const int nloc = nt * 16 + (lane & 15);

  for (int i = tid; i < Bz * Hz; i += 1024) hb[i] = (bf16)0.f;

  float c[8];
#pragma unroll
  for (int r = 0; r < 8; ++r) c[r] = 0.f;

  float bsum[4];
#pragma unroll
  for (int q = 0; q < 4; ++q) bsum[q] = bih[q * Hz + nloc] + bhh[q * Hz + nloc];

  const int xrs = Sz * DIN;  // batch-row stride of x[B,S,DIN]
  __syncthreads();

  for (int t = 0; t < Sz; ++t) {
    v8f acc[4];
#pragma unroll
    for (int q = 0; q < 4; ++q)
#pragma unroll
      for (int r = 0; r < 8; ++r) acc[q][r] = bsum[q];

    // input contribution: A = x_t rows [mt*16 .. mt*16+15]
    const bf16* xbase = x + (size_t)(mt * 16) * xrs + t * DIN;
#pragma unroll
    for (int k0 = 0; k0 < DIN; k0 += 32) {
      v16bf a = load_a_frag(xbase + k0, xrs, lane);
#pragma unroll
      for (int q = 0; q < 4; ++q) {
        v16bf bm = load_b_frag(Wih + k0, DIN, q * Hz + nt * 16, lane);
        acc[q] = __builtin_amdgcn_wmma_f32_16x16x32_bf16(false, a, false, bm,
                                                         (short)0, acc[q], false, false);
      }
    }
    // recurrent contribution: A = h_{t-1} from LDS
#pragma unroll
    for (int k0 = 0; k0 < Hz; k0 += 32) {
      v16bf a = load_a_frag(hb + (mt * 16) * Hz + k0, Hz, lane);
#pragma unroll
      for (int q = 0; q < 4; ++q) {
        v16bf bm = load_b_frag(Whh + k0, Hz, q * Hz + nt * 16, lane);
        acc[q] = __builtin_amdgcn_wmma_f32_16x16x32_bf16(false, a, false, bm,
                                                         (short)0, acc[q], false, false);
      }
    }
    __syncthreads();  // all reads of previous h complete

#pragma unroll
    for (int r = 0; r < 8; ++r) {
      int m = mt * 16 + half * 8 + r;
      float ig = sigf(acc[0][r]);
      float fg = sigf(acc[1][r]);
      float gg = tanhf(acc[2][r]);
      float og = sigf(acc[3][r]);
      float cn = fg * c[r] + ig * gg;
      c[r] = cn;
      float hn = og * tanhf(cn);
      bf16 hv = (bf16)hn;
      hb[m * Hz + nloc] = hv;
      if (write_seq) {
        hseq[((size_t)m * Sz + t) * Hz + nloc] = hv;
      } else if (t == last_idx[m]) {
        outb[m * Hz + nloc] = hv;
      }
    }
    __syncthreads();  // new h visible before next step's reads
  }
}

// ---------------------------------------------------------------------------
// feat = outB[32,256] @ Wlin^T[256,12800] + blin, scattered into padded
// featp[B, NUP, E] (rows u>=100 stay zero).
// ---------------------------------------------------------------------------
__global__ __launch_bounds__(256) void feat_gemm_kernel(
    const bf16* __restrict__ outb, const bf16* __restrict__ Wlin,
    const float* __restrict__ blin, bf16* __restrict__ featp) {
  int lane = threadIdx.x & 31, w = threadIdx.x >> 5;
  int mt = w & 1;
  int n0 = (blockIdx.x * 4 + (w >> 1)) * 16;
  int half = lane >> 4;
  int col = n0 + (lane & 15);
  v8f acc;
  float bv = blin[col];
#pragma unroll
  for (int r = 0; r < 8; ++r) acc[r] = bv;
#pragma unroll
  for (int k0 = 0; k0 < Hz; k0 += 32) {
    v16bf a  = load_a_frag(outb + mt * 16 * Hz + k0, Hz, lane);
    v16bf bm = load_b_frag(Wlin + k0, Hz, n0, lane);
    acc = __builtin_amdgcn_wmma_f32_16x16x32_bf16(false, a, false, bm,
                                                  (short)0, acc, false, false);
  }
  int u = col >> 7, e = col & 127;  // j -> (u, e), Ez == 128
#pragma unroll
  for (int r = 0; r < 8; ++r) {
    int bi = mt * 16 + half * 8 + r;
    featp[((size_t)bi * NUP + u) * Ez + e] = (bf16)acc[r];
  }
}

// ---------------------------------------------------------------------------
// scores[b,v] = max_u featp[b,u,:] . vemb[v,:]   (dominant 41-GFLOP GEMM)
// featp[b] (28 KB) is staged into LDS once per block by the Tensor Data Mover
// (tensor_load_to_lds, TENSORcnt), then A-fragments come from LDS and each
// wave computes two 16-column tiles (A reuse + dual accumulation chains).
// Block covers 256 V columns; grid = (196, 32).
// ---------------------------------------------------------------------------
__global__ __launch_bounds__(256) void score_max_kernel(
    const bf16* __restrict__ featp, const bf16* __restrict__ vembb,
    float* __restrict__ scores) {
  __shared__ bf16 fs[NUP * Ez];   // 28672 bytes
  int b = blockIdx.y;

  // --- TDM: DMA featp[b] -> LDS as one 7168-dword (data_size=4B) row tile ---
  if (threadIdx.x < 32) {
    unsigned long long ga = (unsigned long long)(const void*)(featp + (size_t)b * NUP * Ez);
    unsigned lds = (unsigned)(unsigned long long)(void*)fs;  // LDS byte offset
    const unsigned NDW = (NUP * Ez * 2) / 4;  // 7168 dwords
    v4u g0;
    g0[0] = 1u;                                   // count=1, is_restore=0, gather=0
    g0[1] = lds;                                  // lds_addr
    g0[2] = (unsigned)ga;                         // global_addr[31:0]
    g0[3] = (unsigned)((ga >> 32) & 0x01FFFFFFu)  // global_addr[56:32]
            | (2u << 30);                         // type=2 ("image")
    v8i g1;
    g1[0] = (int)(2u << 16);                      // data_size=2 -> 4 bytes
    g1[1] = (int)((NDW & 0xFFFFu) << 16);         // tensor_dim0[15:0]
    g1[2] = (int)(((NDW >> 16) & 0xFFFFu) | (1u << 16)); // dim0[31:16], tensor_dim1=1
    g1[3] = (int)(NDW << 16);                     // tensor_dim1 hi=0, tile_dim0=NDW
    g1[4] = (int)1;                               // tile_dim1=1, tile_dim2=0
    g1[5] = (int)NDW;                             // tensor_dim0_stride[31:0]
    g1[6] = 0;                                    // stride0 hi, stride1 lo
    g1[7] = 0;
    v4i g2; g2[0] = 0; g2[1] = 0; g2[2] = 0; g2[3] = 0;
    v4i g3; g3[0] = 0; g3[1] = 0; g3[2] = 0; g3[3] = 0;
    v8i g4; g4[0] = 0; g4[1] = 0; g4[2] = 0; g4[3] = 0;
    g4[4] = 0; g4[5] = 0; g4[6] = 0; g4[7] = 0;
    __builtin_amdgcn_tensor_load_to_lds(g0, g1, g2, g3, g4, 0);
    __builtin_amdgcn_s_wait_tensorcnt(0);
  }
  __syncthreads();

  int lane = threadIdx.x & 31, w = threadIdx.x >> 5;
  int half = lane >> 4;
  int nbase = blockIdx.x * 256 + w * 32;
  int ncol0 = nbase + (lane & 15);
  int ncol1 = nbase + 16 + (lane & 15);
  int nc0 = ncol0 < Vz ? ncol0 : (Vz - 1);   // clamp keeps EXEC uniform
  int nc1 = ncol1 < Vz ? ncol1 : (Vz - 1);
  float max0 = -3.402823466e38f, max1 = -3.402823466e38f;

#pragma unroll
  for (int ut = 0; ut < 7; ++ut) {
    v8f acc0, acc1;
#pragma unroll
    for (int r = 0; r < 8; ++r) { acc0[r] = 0.f; acc1[r] = 0.f; }
#pragma unroll
    for (int k0 = 0; k0 < Ez; k0 += 32) {
      v16bf a  = load_a_frag(fs + ut * 16 * Ez + k0, Ez, lane);   // from LDS
      v16bf b0 = *(const v16bf*)(vembb + (size_t)nc0 * Ez + k0 + half * 16);
      v16bf b1 = *(const v16bf*)(vembb + (size_t)nc1 * Ez + k0 + half * 16);
      acc0 = __builtin_amdgcn_wmma_f32_16x16x32_bf16(false, a, false, b0,
                                                     (short)0, acc0, false, false);
      acc1 = __builtin_amdgcn_wmma_f32_16x16x32_bf16(false, a, false, b1,
                                                     (short)0, acc1, false, false);
    }
#pragma unroll
    for (int r = 0; r < 8; ++r) {
      int u = ut * 16 + half * 8 + r;
      if (u < NUSz) {
        max0 = fmaxf(max0, acc0[r]);
        max1 = fmaxf(max1, acc1[r]);
      }
    }
  }
  max0 = fmaxf(max0, __shfl_xor(max0, 16, 32));  // combine the two lane halves
  max1 = fmaxf(max1, __shfl_xor(max1, 16, 32));
  if (half == 0) {
    if (ncol0 < Vz) scores[(size_t)b * Vz + ncol0] = max0;
    if (ncol1 < Vz) scores[(size_t)b * Vz + ncol1] = max1;
  }
}

// ---------------------------------------------------------------------------
// log-softmax statistics + masked NLL loss per batch row
// ---------------------------------------------------------------------------
__global__ __launch_bounds__(256) void loss_kernel(
    const float* __restrict__ scores, const int* __restrict__ label,
    const int* __restrict__ label_type, const int* __restrict__ last_idx,
    float* __restrict__ out) {
  __shared__ float red[256];
  int b = blockIdx.x, tid = threadIdx.x;
  const float* srow = scores + (size_t)b * Vz;
  float m = -3.402823466e38f;
  for (int v = tid; v < Vz; v += 256) m = fmaxf(m, srow[v]);
  red[tid] = m; __syncthreads();
  for (int s = 128; s > 0; s >>= 1) {
    if (tid < s) red[tid] = fmaxf(red[tid], red[tid + s]);
    __syncthreads();
  }
  float M = red[0]; __syncthreads();
  float z = 0.f;
  for (int v = tid; v < Vz; v += 256) z += __expf(srow[v] - M);
  red[tid] = z; __syncthreads();
  for (int s = 128; s > 0; s >>= 1) {
    if (tid < s) red[tid] += red[tid + s];
    __syncthreads();
  }
  if (tid == 0) {
    float logZ = __logf(red[0]);
    int li = last_idx[b];
    const int* lab = label      + ((size_t)b * Sz + li) * Lz;
    const int* typ = label_type + ((size_t)b * Sz + li) * Lz;
    float s = 0.f;
    for (int j = 0; j < Lz; ++j) s += (float)typ[j] * (srow[lab[j]] - M - logZ);
    float loss = -0.5f * s / (float)Lz;
    out[b] = loss; out[Bz + b] = loss;
  }
}

// ---------------------------------------------------------------------------
// top-k membership via exact rank counting (matches jax.lax.top_k tie-breaking):
// label in top-num_rec  <=>  #{s > s_lab} + #{s == s_lab, idx < lab} < num_rec
// ---------------------------------------------------------------------------
__global__ __launch_bounds__(256) void hit_kernel(
    const float* __restrict__ scores, const int* __restrict__ label,
    const int* __restrict__ label_type, const int* __restrict__ last_idx,
    int* __restrict__ counters) {
  __shared__ int   lab_s[Lz];
  __shared__ float slab_s[Lz];
  __shared__ int   C1[Lz], C2[Lz];
  __shared__ int   nrec_s;
  int b = blockIdx.x, tid = threadIdx.x;
  const float* srow = scores + (size_t)b * Vz;
  if (tid == 0) {
    int li = last_idx[b];
    const int* lab = label      + ((size_t)b * Sz + li) * Lz;
    const int* typ = label_type + ((size_t)b * Sz + li) * Lz;
    int nr = 0;
    for (int j = 0; j < Lz; ++j) {
      lab_s[j]  = lab[j];
      slab_s[j] = srow[lab[j]];
      C1[j] = 0; C2[j] = 0;
      nr += typ[j];
    }
    nrec_s = nr;
  }
  __syncthreads();
  int c1[Lz], c2[Lz];
#pragma unroll
  for (int j = 0; j < Lz; ++j) { c1[j] = 0; c2[j] = 0; }
  for (int v = tid; v < Vz; v += 256) {
    float s = srow[v];
#pragma unroll
    for (int j = 0; j < Lz; ++j) {
      c1[j] += (s > slab_s[j]) ? 1 : 0;
      c2[j] += (s == slab_s[j] && v < lab_s[j]) ? 1 : 0;
    }
  }
  for (int j = 0; j < Lz; ++j) { atomicAdd(&C1[j], c1[j]); atomicAdd(&C2[j], c2[j]); }
  __syncthreads();
  if (tid == 0) {
    int nr = nrec_s, acc = 0;
    for (int j = 0; j < nr; ++j)
      if (C1[j] + C2[j] < nr) acc++;
    atomicAdd(&counters[0], acc);
    atomicAdd(&counters[1], nr);
  }
}

__global__ void finalize_kernel(const int* __restrict__ counters, float* __restrict__ out) {
  if (threadIdx.x == 0) {
    float cnt = (float)counters[1];
    out[64] = (float)counters[0] / cnt;
    out[65] = cnt;
  }
}

// ---------------------------------------------------------------------------
// host side
// ---------------------------------------------------------------------------
extern "C" void kernel_launch(void* const* d_in, const int* in_sizes, int n_in,
                              void* d_out, int out_size, void* d_ws, size_t ws_size,
                              hipStream_t stream) {
  (void)in_sizes; (void)n_in; (void)out_size; (void)ws_size;
  const int*   inputs     = (const int*)  d_in[0];
  const int*   label      = (const int*)  d_in[1];
  const int*   label_type = (const int*)  d_in[2];
  const int*   mask       = (const int*)  d_in[3];
  const float* emb_table  = (const float*)d_in[4];
  const float* vemb       = (const float*)d_in[5];
  const float* Wih0 = (const float*)d_in[6];
  const float* Whh0 = (const float*)d_in[7];
  const float* bih0 = (const float*)d_in[8];
  const float* bhh0 = (const float*)d_in[9];
  const float* Wih1 = (const float*)d_in[10];
  const float* Whh1 = (const float*)d_in[11];
  const float* bih1 = (const float*)d_in[12];
  const float* bhh1 = (const float*)d_in[13];
  const float* Wlin = (const float*)d_in[14];
  const float* blin = (const float*)d_in[15];
  float* out = (float*)d_out;

  char* ws = (char*)d_ws;
  size_t off = 0;
  auto alloc = [&](size_t nbytes) -> void* {
    off = (off + 255) & ~(size_t)255;
    void* p = ws + off;
    off += nbytes;
    return p;
  };
  bf16* vembb  = (bf16*)alloc((size_t)Vz * Ez * 2);
  bf16* Wih0b  = (bf16*)alloc((size_t)4 * Hz * Ez * 2);
  bf16* Whh0b  = (bf16*)alloc((size_t)4 * Hz * Hz * 2);
  bf16* Wih1b  = (bf16*)alloc((size_t)4 * Hz * Hz * 2);
  bf16* Whh1b  = (bf16*)alloc((size_t)4 * Hz * Hz * 2);
  bf16* Wlinb  = (bf16*)alloc((size_t)Ez * NUSz * Hz * 2);
  bf16* xb     = (bf16*)alloc((size_t)Bz * Sz * Ez * 2);
  bf16* h0b    = (bf16*)alloc((size_t)Bz * Sz * Hz * 2);
  bf16* outbb  = (bf16*)alloc((size_t)Bz * Hz * 2);
  bf16* featp  = (bf16*)alloc((size_t)Bz * NUP * Ez * 2);
  float* scores = (float*)alloc((size_t)Bz * Vz * 4);
  int*  last_idx = (int*)alloc(Bz * 4);
  int*  counters = (int*)alloc(2 * 4);

  // weight / embedding down-conversion to bf16
  cvt_bf16_kernel<<<1024, 256, 0, stream>>>(vemb, vembb, Vz * Ez);
  cvt_bf16_kernel<<<256, 256, 0, stream>>>(Wih0, Wih0b, 4 * Hz * Ez);
  cvt_bf16_kernel<<<256, 256, 0, stream>>>(Whh0, Whh0b, 4 * Hz * Hz);
  cvt_bf16_kernel<<<256, 256, 0, stream>>>(Wih1, Wih1b, 4 * Hz * Hz);
  cvt_bf16_kernel<<<256, 256, 0, stream>>>(Whh1, Whh1b, 4 * Hz * Hz);
  cvt_bf16_kernel<<<1024, 256, 0, stream>>>(Wlin, Wlinb, Ez * NUSz * Hz);

  util_init_kernel<<<1, 64, 0, stream>>>(mask, last_idx, counters);
  zero_bf16_kernel<<<256, 256, 0, stream>>>(featp, Bz * NUP * Ez);
  embed_kernel<<<256, 256, 0, stream>>>(inputs, mask, emb_table, xb);

  lstm_layer_kernel<Ez><<<1, 1024, 0, stream>>>(xb, Wih0b, Whh0b, bih0, bhh0,
                                                h0b, (bf16*)nullptr, last_idx, 1);
  lstm_layer_kernel<Hz><<<1, 1024, 0, stream>>>(h0b, Wih1b, Whh1b, bih1, bhh1,
                                                (bf16*)nullptr, outbb, last_idx, 0);

  feat_gemm_kernel<<<200, 256, 0, stream>>>(outbb, Wlinb, blin, featp);

  dim3 sg((Vz + 255) / 256, Bz);
  score_max_kernel<<<sg, 256, 0, stream>>>(featp, vembb, scores);

  loss_kernel<<<Bz, 256, 0, stream>>>(scores, label, label_type, last_idx, out);
  hit_kernel<<<Bz, 256, 0, stream>>>(scores, label, label_type, last_idx, counters);
  finalize_kernel<<<1, 64, 0, stream>>>(counters, out);
}
